// MinGRU_69329362092185
// MI455X (gfx1250) — matrix operand order repaired and checked by reading, not compiled, not run on hardware
//
#include <hip/hip_runtime.h>
#include <hip/hip_bf16.h>
#include <math.h>

// ---------------------------------------------------------------------------
// CDNA5 WMMA types (probe-confirmed):
//   v8f = __builtin_amdgcn_wmma_f32_16x16x32_bf16(bool, v16bf, bool, v16bf,
//                                                 short, v8f, bool, bool)
// ---------------------------------------------------------------------------
typedef __attribute__((ext_vector_type(16))) __bf16 v16bf;
typedef __attribute__((ext_vector_type(8)))  float  v8f;

struct alignas(16) F4 { float x, y, z, w; };

__device__ __forceinline__ v8f wmma_bf16(v16bf a, v16bf b, v8f c) {
    return __builtin_amdgcn_wmma_f32_16x16x32_bf16(false, a, false, b,
                                                   (short)0, c, false, false);
}

__device__ __forceinline__ void cvt8(v16bf& dst, int base, const F4& a, const F4& b) {
    dst[base + 0] = (__bf16)a.x; dst[base + 1] = (__bf16)a.y;
    dst[base + 2] = (__bf16)a.z; dst[base + 3] = (__bf16)a.w;
    dst[base + 4] = (__bf16)b.x; dst[base + 5] = (__bf16)b.y;
    dst[base + 6] = (__bf16)b.z; dst[base + 7] = (__bf16)b.w;
}

// B fragment (32x16 bf16) gathered straight from a weight row (N x K, row-major).
// lane L holds col N=L&15; L<16 -> K 0..15, L>=16 -> K 16..31 (q pre-offset).
__device__ __forceinline__ v16bf load_bfrag(const float* __restrict__ q) {
    F4 b0 = *(const F4*)(q);      F4 b1 = *(const F4*)(q + 4);
    F4 b2 = *(const F4*)(q + 8);  F4 b3 = *(const F4*)(q + 12);
    v16bf bf;
    cvt8(bf, 0, b0, b1);
    cvt8(bf, 8, b2, b3);
    return bf;
}

// A fragment (16x32 bf16) gathered from LDS-staged chunk (row-major 16x32,
// packed as 16x16 dwords). lane L: row M=L&15; L<16: K 0..15;
// L>=16: K 16..31. Two ds_load_b128 per lane.
__device__ __forceinline__ v16bf load_afrag_lds(const uint32_t* As32, int lane) {
    const int r  = lane & 15;
    const int ao = (lane & 16) ? 8 : 0;  // dword offset = aoff/2
    union { uint4 q[2]; v16bf v; } u;
    u.q[0] = *(const uint4*)(As32 + r * 16 + ao);
    u.q[1] = *(const uint4*)(As32 + r * 16 + 4 + ao);
    return u.v;
}

__device__ __forceinline__ float laexp(float x, float y) {
    float m = fmaxf(x, y);
    if (m == -INFINITY) return -INFINITY;
    return m + log1pf(__expf(-fabsf(x - y)));
}

// ---------------------------------------------------------------------------
// GEMM1 fused: for each (m, h) compute hidden & gate (paired N-tiles n, n+H),
// then the elementwise log-space quantities:
//   a = log_coeffs = -softplus(gate)
//   v = log_values = (gate - softplus(gate)) + log_g(hidden)
// A chunk staged through LDS (bf16), shared by all 8 waves of the block.
// ---------------------------------------------------------------------------
__global__ void mingru_gemm1_fused(const float* __restrict__ A,    // M x K
                                   const float* __restrict__ Whg,  // 2H x K
                                   float* __restrict__ loga,       // M x H
                                   float* __restrict__ logv,       // M x H
                                   int H, int K)
{
    __shared__ __align__(16) uint32_t As32[16 * 16];   // 16x32 bf16 chunk

    const int lane    = threadIdx.x & 31;
    const int wave    = threadIdx.x >> 5;
    const int groupsN = (H / 16) / 8;                  // 8
    const int tileM   = blockIdx.x / groupsN;
    const int tileN   = (blockIdx.x % groupsN) * 8 + wave;

    const int col  = tileN * 16 + (lane & 15);
    const int boff = (lane & 16) ? 16 : 0;
    const float* __restrict__ wrow_h = Whg + (size_t)col * K;
    const float* __restrict__ wrow_g = Whg + (size_t)(col + H) * K;

    // staging assignment: thread p owns (srow, dword skp) of the LDS chunk
    const int p = threadIdx.x, srow = p >> 4, skp = p & 15;
    const float* __restrict__ grow =
        A + (size_t)(tileM * 16 + srow) * K + skp * 2;

    v8f acc_h = {}, acc_g = {};
    for (int kc = 0; kc < K; kc += 32) {
        __builtin_prefetch(wrow_h + kc + 32, 0, 1);
        __builtin_prefetch(wrow_g + kc + 32, 0, 1);
        {   // cooperative stage: f32 -> bf16 pair -> LDS dword
            float2 f = *(const float2*)(grow + kc);
            union { __bf16 h2[2]; uint32_t u; } pk;
            pk.h2[0] = (__bf16)f.x; pk.h2[1] = (__bf16)f.y;
            As32[srow * 16 + skp] = pk.u;
        }
        __syncthreads();
        v16bf af = load_afrag_lds(As32, lane);
        v16bf bh = load_bfrag(wrow_h + kc + boff);
        v16bf bg = load_bfrag(wrow_g + kc + boff);
        acc_h = wmma_bf16(af, bh, acc_h);
        acc_g = wmma_bf16(af, bg, acc_g);
        __syncthreads();
    }

    const int mbase = (lane & 16) ? 8 : 0;
#pragma unroll
    for (int r = 0; r < 8; ++r) {
        int   m   = tileM * 16 + mbase + r;
        float hid = acc_h[r], gat = acc_g[r];
        float sp_g = fmaxf(gat, 0.0f) + log1pf(__expf(-fabsf(gat)));
        float a  = -sp_g;
        float lz = gat - sp_g;
        float lg = (hid >= 0.0f)
                 ? __logf(hid + 0.5f)
                 : -(fmaxf(-hid, 0.0f) + log1pf(__expf(-fabsf(hid))));
        size_t off = (size_t)m * H + col;
        loga[off] = a;
        logv[off] = lz + lg;
    }
}

// ---------------------------------------------------------------------------
// Chunked associative scan (3 passes). Elements e_t = (a_t, v_t, beta_t),
// combine per reference. T split into Cn chunks of L.
// ---------------------------------------------------------------------------
__global__ void scan_chunk_reduce(const float* __restrict__ loga,
                                  const float* __restrict__ logv,
                                  const float* __restrict__ is_init,
                                  float* __restrict__ sumA,
                                  float* __restrict__ sumB,
                                  float* __restrict__ sumb,
                                  int B, int T, int H, int Cn, int L)
{
    int idx = blockIdx.x * blockDim.x + threadIdx.x;
    if (idx >= B * Cn * H) return;
    int hh  = idx % H;
    int rem = idx / H;
    int c   = rem % Cn;
    int b   = rem / Cn;

    const int t0 = c * L;
    size_t off0  = ((size_t)b * T + t0) * H + hh;

    // state := first element of the chunk
    float la = loga[off0];
    float lb = logv[off0];
    float bb = fminf(is_init[(size_t)b * T + t0], 1.0f);

    for (int i = 1; i < L; ++i) {
        size_t off = off0 + (size_t)i * H;
        float a    = loga[off];
        float v    = logv[off];
        float beta = is_init[(size_t)b * T + t0 + i];
        float lb2  = laexp(a + lb, v);
        la = a   * (1.0f - beta) + la * beta;
        lb = lb2 * (1.0f - beta) + v  * beta;
        bb = fminf(bb + beta, 1.0f);
    }
    size_t so = ((size_t)b * Cn + c) * H + hh;
    sumA[so] = la; sumB[so] = lb; sumb[so] = bb;
}

__global__ void scan_prefix(const float* __restrict__ sumA,
                            const float* __restrict__ sumB,
                            const float* __restrict__ sumb,
                            float* __restrict__ preA,
                            float* __restrict__ preB,
                            int B, int H, int Cn)
{
    int idx = blockIdx.x * blockDim.x + threadIdx.x;
    if (idx >= B * H) return;
    int hh = idx % H;
    int b  = idx / H;

    float la = 0.0f, lb = -INFINITY;
    for (int c = 0; c < Cn; ++c) {
        size_t so = ((size_t)b * Cn + c) * H + hh;
        preA[so] = la;                         // exclusive prefix
        preB[so] = lb;
        float A2 = sumA[so], B2 = sumB[so], b2 = sumb[so];
        float lbn = laexp(A2 + lb, B2);
        la = A2  * (1.0f - b2) + la * b2;
        lb = lbn * (1.0f - b2) + B2 * b2;
    }
}

__global__ void scan_apply(const float* __restrict__ loga,
                           const float* __restrict__ logv,
                           const float* __restrict__ is_init,
                           const float* __restrict__ preA,
                           const float* __restrict__ preB,
                           const float* __restrict__ h0,
                           float* __restrict__ h,
                           float* __restrict__ h_n,
                           int B, int T, int H, int Cn, int L)
{
    int idx = blockIdx.x * blockDim.x + threadIdx.x;
    if (idx >= B * Cn * H) return;
    int hh  = idx % H;
    int rem = idx / H;
    int c   = rem % Cn;
    int b   = rem / Cn;

    size_t so = ((size_t)b * Cn + c) * H + hh;
    float la  = preA[so];
    float lb  = preB[so];
    float lh0 = __logf(h0[(size_t)b * H + hh]);

    const int t0 = c * L;
    size_t off0  = ((size_t)b * T + t0) * H + hh;
    float ht = 0.0f;

    for (int i = 0; i < L; ++i) {
        size_t off = off0 + (size_t)i * H;
        float a    = loga[off];
        float v    = logv[off];
        float beta = is_init[(size_t)b * T + t0 + i];
        float lb2  = laexp(a + lb, v);
        la = a   * (1.0f - beta) + la * beta;
        lb = lb2 * (1.0f - beta) + v  * beta;
        ht = __expf(laexp(la + lh0, lb));
        h[off] = ht;
    }
    if (c == Cn - 1)
        h_n[(size_t)b * H + hh] = ht;
}

// ---------------------------------------------------------------------------
// GEMM2: out = h * W_out^T. One block per 16-row M tile covering all of
// N = 512: 8 waves x 4 N-tiles, 4 WMMAs per K-chunk off one LDS A fragment.
// ---------------------------------------------------------------------------
__global__ void mingru_gemm2(const float* __restrict__ Hm,    // M x K
                             const float* __restrict__ Wout,  // N x K
                             float* __restrict__ Out,         // M x N
                             int N, int K)
{
    __shared__ __align__(16) uint32_t As32[16 * 16];

    const int lane  = threadIdx.x & 31;
    const int wave  = threadIdx.x >> 5;
    const int tileM = blockIdx.x;
    const int boff  = (lane & 16) ? 16 : 0;

    int colv[4];
    const float* wrow[4];
#pragma unroll
    for (int j = 0; j < 4; ++j) {
        int tileN = wave * 4 + j;
        colv[j] = tileN * 16 + (lane & 15);
        wrow[j] = Wout + (size_t)colv[j] * K;
    }

    const int p = threadIdx.x, srow = p >> 4, skp = p & 15;
    const float* __restrict__ grow =
        Hm + (size_t)(tileM * 16 + srow) * K + skp * 2;

    v8f acc[4] = {};
    for (int kc = 0; kc < K; kc += 32) {
        {
            float2 f = *(const float2*)(grow + kc);
            union { __bf16 h2[2]; uint32_t u; } pk;
            pk.h2[0] = (__bf16)f.x; pk.h2[1] = (__bf16)f.y;
            As32[srow * 16 + skp] = pk.u;
        }
        __syncthreads();
        v16bf af = load_afrag_lds(As32, lane);
#pragma unroll
        for (int j = 0; j < 4; ++j) {
            v16bf bf = load_bfrag(wrow[j] + kc + boff);
            acc[j] = wmma_bf16(af, bf, acc[j]);
        }
        __syncthreads();
    }

    const int mbase = (lane & 16) ? 8 : 0;
#pragma unroll
    for (int j = 0; j < 4; ++j) {
        float* crow = Out + (size_t)(tileM * 16) * N + colv[j];
#pragma unroll
        for (int r = 0; r < 8; ++r)
            crow[(size_t)(mbase + r) * N] = acc[j][r];
    }
}

// ---------------------------------------------------------------------------
// Launcher
// ---------------------------------------------------------------------------
extern "C" void kernel_launch(void* const* d_in, const int* in_sizes, int n_in,
                              void* d_out, int out_size, void* d_ws, size_t ws_size,
                              hipStream_t stream)
{
    const float* input   = (const float*)d_in[0]; // (B,T,Din)
    const float* is_init = (const float*)d_in[1]; // (B,T,1)
    const float* h0      = (const float*)d_in[2]; // (B,1,H)
    const float* W_hg    = (const float*)d_in[3]; // (2H,Din)
    const float* W_out   = (const float*)d_in[4]; // (Dout,H)

    const int B = 8, T = 4096, Din = 512, H = 1024, Dout = 512;
    const int M  = B * T;       // 32768
    const int Cn = 16;          // scan chunks
    const int L  = T / Cn;      // 256

    // workspace layout (floats)
    float* loga = (float*)d_ws;                    // M*H
    float* logv = loga + (size_t)M * H;            // M*H
    float* h    = logv + (size_t)M * H;            // M*H
    float* sumA = h    + (size_t)M * H;            // B*Cn*H
    float* sumB = sumA + (size_t)B * Cn * H;
    float* sumb = sumB + (size_t)B * Cn * H;
    float* preA = sumb + (size_t)B * Cn * H;
    float* preB = preA + (size_t)B * Cn * H;

    float* out = (float*)d_out;                    // M x Dout
    float* h_n = out + (size_t)M * Dout;           // B x H

    {   // GEMM1 fused (hidden+gate pairs -> loga/logv)
        int blocks = (M / 16) * ((H / 16) / 8);    // 2048 * 8 = 16384
        mingru_gemm1_fused<<<blocks, 256, 0, stream>>>(
            input, W_hg, loga, logv, H, Din);
    }
    {   // scan pass 1: per-chunk reduction
        int total = B * Cn * H;                    // 131072
        scan_chunk_reduce<<<(total + 255) / 256, 256, 0, stream>>>(
            loga, logv, is_init, sumA, sumB, sumb, B, T, H, Cn, L);
    }
    {   // scan pass 2: exclusive prefix over chunks
        int total = B * H;                         // 8192
        scan_prefix<<<(total + 255) / 256, 256, 0, stream>>>(
            sumA, sumB, sumb, preA, preB, B, H, Cn);
    }
    {   // scan pass 3: replay with incoming state, emit h / h_n
        int total = B * Cn * H;
        scan_apply<<<(total + 255) / 256, 256, 0, stream>>>(
            loga, logv, is_init, preA, preB, h0, h, h_n, B, T, H, Cn, L);
    }
    {   // GEMM2: out = h * W_out^T
        int blocks = M / 16;                       // 2048
        mingru_gemm2<<<blocks, 256, 0, stream>>>(h, W_out, out, Dout, H);
    }
}
